// BertWithGAP_60275571032232
// MI455X (gfx1250) — matrix operand order repaired and checked by reading, not compiled, not run on hardware
//
#include <hip/hip_runtime.h>

// ---------------------------------------------------------------------------
// BertWithGAP on MI455X (gfx1250): 3x basis-RGCN (bf16 WMMA GEMMs + CSR
// gather aggregation -- no float atomics), encoder dense via WMMA, fused
// gather/mean/gf, block-per-graph attention pooling + MLP head.
// ---------------------------------------------------------------------------

typedef __attribute__((ext_vector_type(16))) __bf16       v16bf;
typedef __attribute__((ext_vector_type(8)))  float        v8f;
typedef __attribute__((ext_vector_type(4)))  unsigned int u32x4;
typedef unsigned short u16;

// problem constants (match reference)
constexpr int B_   = 32;
constexpr int N_   = 512;
constexpr int K_   = 4;
constexpr int E_   = 524288;
constexpr int R_   = 12;
constexpr int NB_  = 12;
constexpr int IN_  = 200;
constexpr int H_   = 256;
constexpr int OUT_ = 200;
constexpr int HS_  = 768;
constexpr int L_   = 512;
constexpr int NN_  = B_ * N_;
constexpr int INP_  = 224;  // IN  padded to multiple of 32 (WMMA K)
constexpr int OUTP_ = 208;  // OUT padded to multiple of 16 (WMMA N)

#define DEVI __device__ __forceinline__

DEVI u16 f2bf(float f) {
  unsigned int u = __float_as_uint(f);
  unsigned int r = u + 0x7FFFu + ((u >> 16) & 1u);  // RTNE
  return (u16)(r >> 16);
}
DEVI float bflo(unsigned int w) { return __uint_as_float(w << 16); }
DEVI float bfhi(unsigned int w) { return __uint_as_float(w & 0xFFFF0000u); }

// ---------------------------------------------------------------------------
// Weight prep: Wt[r][n][k] = sum_b comp[r,b] * V[b,k,n], bf16, transposed+padded
// ---------------------------------------------------------------------------
__global__ void prep_w_kernel(const float* __restrict__ V, const float* __restrict__ comp,
                              u16* __restrict__ Wt, int Kdim, int Ndim, int Kpad, int Npad) {
  int i = blockIdx.x * blockDim.x + threadIdx.x;
  int total = R_ * Ndim * Kdim;
  if (i >= total) return;
  int k = i % Kdim;
  int n = (i / Kdim) % Ndim;
  int r = i / (Kdim * Ndim);
  float a = 0.f;
#pragma unroll
  for (int b = 0; b < NB_; ++b)
    a += comp[r * NB_ + b] * V[((long)b * Kdim + k) * Ndim + n];
  Wt[(long)r * Npad * Kpad + (long)n * Kpad + k] = f2bf(a);
}

// Wdt[n][k] = Wd[k][n]  (bf16, [OUTP_ x HS_], pad rows pre-zeroed)
__global__ void prep_wd_kernel(const float* __restrict__ Wd, u16* __restrict__ Wdt) {
  int i = blockIdx.x * blockDim.x + threadIdx.x;
  if (i >= OUT_ * HS_) return;
  int k = i % HS_;
  int n = i / HS_;
  Wdt[(long)n * HS_ + k] = f2bf(Wd[(long)k * OUT_ + n]);
}

// fp32 -> bf16 with column padding (pad region pre-zeroed by memset)
__global__ void conv_pad_kernel(const float* __restrict__ X, u16* __restrict__ Y,
                                int total, int Kdim, int Kpad) {
  int i = blockIdx.x * blockDim.x + threadIdx.x;
  if (i >= total) return;
  int r = i / Kdim;
  int k = i - r * Kdim;
  Y[(long)r * Kpad + k] = f2bf(X[i]);
}

// h_next = bf16(relu(agg + bias))   (Kdim == Kpad == 256 here)
__global__ void bias_relu_bf16_kernel(const float* __restrict__ agg, const float* __restrict__ bias,
                                      u16* __restrict__ Y, int total, int Kdim) {
  int i = blockIdx.x * blockDim.x + threadIdx.x;
  if (i >= total) return;
  int k = i % Kdim;
  float v = agg[i] + bias[k];
  Y[i] = f2bf(v > 0.f ? v : 0.f);
}

// ---------------------------------------------------------------------------
// CSR construction over dst (built once per launch, reused by all 3 layers)
// ---------------------------------------------------------------------------
__global__ void deg_kernel(const int* __restrict__ dst, int* __restrict__ deg) {
  int e = blockIdx.x * blockDim.x + threadIdx.x;
  if (e < E_) atomicAdd(&deg[dst[e]], 1);
}

// single block, 256 threads, each thread scans 64 contiguous nodes (NN_=16384)
__global__ void scan_kernel(const int* __restrict__ deg, int* __restrict__ offs) {
  __shared__ int part[256];
  __shared__ int pexc[256];
  int t = threadIdx.x;
  int base = t * 64;
  int s = 0;
  for (int i = 0; i < 64; ++i) s += deg[base + i];
  part[t] = s;
  __syncthreads();
  if (t == 0) {
    int run = 0;
    for (int i = 0; i < 256; ++i) { pexc[i] = run; run += part[i]; }
  }
  __syncthreads();
  int run = pexc[t];
  for (int i = 0; i < 64; ++i) { offs[base + i] = run; run += deg[base + i]; }
}

__global__ void fill_kernel(const int* __restrict__ dst, const int* __restrict__ offs,
                            int* __restrict__ cur, int* __restrict__ ebuf) {
  int e = blockIdx.x * blockDim.x + threadIdx.x;
  if (e >= E_) return;
  int d = dst[e];
  int p = atomicAdd(&cur[d], 1);
  ebuf[offs[d] + p] = e;
}

// ---------------------------------------------------------------------------
// WMMA GEMM: C[M, Ntot] = A[M, Kpad](bf16) * Bt[Ntot, Kpad](bf16)^T
// One wave per 16x16 C tile; 8 waves (128 rows) per block. f32 accumulate.
// ---------------------------------------------------------------------------
union FragAB { v16bf v; u32x4 q[2]; };

template <bool OUT_BF16>
__global__ void gemm_wmma_kernel(const u16* __restrict__ A, const u16* __restrict__ Bt,
                                 void* __restrict__ C, int Kpad, int Ntot) {
  const int lane = threadIdx.x & 31;
  const int wave = threadIdx.x >> 5;
  const int half = lane >> 4;   // 0: lanes 0-15, 1: lanes 16-31
  const int lm   = lane & 15;
  const long m0 = (long)(blockIdx.x * 8 + wave) * 16;
  const long n0 = (long)blockIdx.y * 16;
  const u16* __restrict__ arow = A  + (m0 + lm) * (long)Kpad;
  const u16* __restrict__ brow = Bt + (n0 + lm) * (long)Kpad;

  v8f acc = {};
  for (int k = 0; k < Kpad; k += 32) {
    FragAB fa, fb;
    // A 16x32 bf16: lanes 0-15 hold K[0..8)+[16..24); lanes 16-31 K[8..16)+[24..32)
    fa.q[0] = *(const u32x4*)(arow + k + half * 8);
    fa.q[1] = *(const u32x4*)(arow + k + 16 + half * 8);
    // B 32x16 bf16 (from K-major Wt): lanes 0-15 col=lm K[0..8)+[16..24); lanes 16-31 K[8..16)+[24..32)
    fb.q[0] = *(const u32x4*)(brow + k + half * 8);
    fb.q[1] = *(const u32x4*)(brow + k + 16 + half * 8);
    acc = __builtin_amdgcn_wmma_f32_16x16x32_bf16(
        /*neg_a=*/false, fa.v, /*neg_b=*/false, fb.v,
        /*c_mod=*/(short)0, acc, /*reuse_a=*/false, /*reuse_b=*/false);
  }

  // C layout: VGPR j -> row (j + 8*half), col = lm
  if constexpr (OUT_BF16) {
    u16* Co = (u16*)C;
#pragma unroll
    for (int j = 0; j < 8; ++j)
      Co[(m0 + half * 8 + j) * (long)Ntot + n0 + lm] = f2bf(acc[j]);
  } else {
    float* Co = (float*)C;
#pragma unroll
    for (int j = 0; j < 8; ++j)
      Co[(m0 + half * 8 + j) * (long)Ntot + n0 + lm] = acc[j];
  }
}

// ---------------------------------------------------------------------------
// CSR gather aggregation: agg[n,:] = sum_{e: dst[e]==n} proj[src[e], etype[e], :]
// One wave per destination node; each lane owns 8 columns in registers.
// No atomics, fully coalesced 16B reads of the bf16 proj rows.
// ---------------------------------------------------------------------------
__global__ void node_agg_kernel(const u16* __restrict__ proj, const int* __restrict__ src,
                                const int* __restrict__ et, const int* __restrict__ offs,
                                const int* __restrict__ deg, const int* __restrict__ ebuf,
                                float* __restrict__ agg, int ldp, int Npad, int Nout) {
  int gw = (int)((blockIdx.x * (long)blockDim.x + threadIdx.x) >> 5);
  int lane = threadIdx.x & 31;
  if (gw >= NN_) return;
  int o0 = lane * 8;
  bool act = (o0 < Nout);
  float acc[8] = {0.f, 0.f, 0.f, 0.f, 0.f, 0.f, 0.f, 0.f};
  int beg = offs[gw];
  int dnum = deg[gw];
  for (int j = 0; j < dnum; ++j) {
    int e = ebuf[beg + j];
    int s = src[e], r = et[e];
    const u16* __restrict__ row = proj + (long)s * ldp + (long)r * Npad;
    if (act) {
      u32x4 q = *(const u32x4*)(row + o0);
      acc[0] += bflo(q[0]); acc[1] += bfhi(q[0]);
      acc[2] += bflo(q[1]); acc[3] += bfhi(q[1]);
      acc[4] += bflo(q[2]); acc[5] += bfhi(q[2]);
      acc[6] += bflo(q[3]); acc[7] += bfhi(q[3]);
    }
  }
  if (act) {
    float* po = agg + (long)gw * Nout + o0;
#pragma unroll
    for (int i = 0; i < 8; ++i) po[i] = acc[i];
  }
}

// ---------------------------------------------------------------------------
// gf[b,n,:] = (agg3 + b3) + masked-mean_k(enc[b, idx_k, :] + bd)  (one wave/node)
// ---------------------------------------------------------------------------
__global__ void gf_fuse_kernel(const float* __restrict__ agg3, const float* __restrict__ b3,
                               const float* __restrict__ enc, const float* __restrict__ bd,
                               const int* __restrict__ midx, const int* __restrict__ mmask,
                               float* __restrict__ gf) {
  int gw = (int)((blockIdx.x * (long)blockDim.x + threadIdx.x) >> 5);
  int lane = threadIdx.x & 31;
  if (gw >= NN_) return;
  int b = gw >> 9;  // N_ = 512
  int idx[K_];
  float msk[K_];
  float cnt = 0.f;
#pragma unroll
  for (int k = 0; k < K_; ++k) {
    idx[k] = midx[(long)gw * K_ + k];
    msk[k] = (mmask[(long)gw * K_ + k] > 0) ? 1.f : 0.f;
    cnt += msk[k];
  }
  float inv = cnt > 0.f ? 1.f / cnt : 0.f;
  const float* __restrict__ eb = enc + (long)b * L_ * OUTP_;
  for (int o = lane; o < OUT_; o += 32) {
    float s = 0.f;
#pragma unroll
    for (int k = 0; k < K_; ++k) s += msk[k] * eb[(long)idx[k] * OUTP_ + o];
    float de = (cnt > 0.f) ? (s * inv + bd[o]) : 0.f;
    gf[(long)gw * OUT_ + o] = agg3[(long)gw * OUT_ + o] + b3[o] + de;
  }
}

// ---------------------------------------------------------------------------
// Per-graph: gate softmax over nodes, attention readout, MLP head, sigmoid.
// One block (512 threads) per graph.
// ---------------------------------------------------------------------------
__global__ void pool_mlp_kernel(const float* __restrict__ gf, const float* __restrict__ Wg,
                                const float* __restrict__ bg, const float* __restrict__ Wm1,
                                const float* __restrict__ bm1, const float* __restrict__ Wm2,
                                const float* __restrict__ bm2, const float* __restrict__ Wm3,
                                const float* __restrict__ bm3, float* __restrict__ out) {
  __shared__ float logit[N_];
  __shared__ float sdata[N_];
  __shared__ float readout[OUT_];
  __shared__ float z1[100];
  __shared__ float z2[64];
  int b = blockIdx.x, t = threadIdx.x;
  const float* __restrict__ g = gf + (long)b * N_ * OUT_;

  // gate logits
  {
    float acc = 0.f;
    const float* row = g + (long)t * OUT_;
    for (int i = 0; i < OUT_; ++i) acc += row[i] * Wg[i];
    logit[t] = acc + bg[0];
    sdata[t] = logit[t];
  }
  __syncthreads();
  for (int s = N_ / 2; s > 0; s >>= 1) {
    if (t < s) sdata[t] = fmaxf(sdata[t], sdata[t + s]);
    __syncthreads();
  }
  float mx = sdata[0];
  __syncthreads();
  float ex = __expf(logit[t] - mx);
  sdata[t] = ex;
  __syncthreads();
  for (int s = N_ / 2; s > 0; s >>= 1) {
    if (t < s) sdata[t] += sdata[t + s];
    __syncthreads();
  }
  float inv = 1.f / sdata[0];
  __syncthreads();
  logit[t] = ex * inv;  // gate
  __syncthreads();

  if (t < OUT_) {
    float r = 0.f;
    for (int n = 0; n < N_; ++n) r += logit[n] * g[(long)n * OUT_ + t];
    readout[t] = r;
  }
  __syncthreads();
  if (t < 100) {
    float a = bm1[t];
    for (int i = 0; i < OUT_; ++i) a += readout[i] * Wm1[i * 100 + t];
    z1[t] = a > 0.f ? a : 0.f;
  }
  __syncthreads();
  if (t < 64) {
    float a = bm2[t];
    for (int i = 0; i < 100; ++i) a += z1[i] * Wm2[i * 64 + t];
    z2[t] = a > 0.f ? a : 0.f;
  }
  __syncthreads();
  if (t == 0) {
    float a = bm3[0];
    for (int i = 0; i < 64; ++i) a += z2[i] * Wm3[i];
    out[b] = 1.f / (1.f + __expf(-a));
  }
}

// ---------------------------------------------------------------------------
// host-side launch
// ---------------------------------------------------------------------------
static inline size_t alignup256(size_t x) { return (x + 255) & ~(size_t)255; }

extern "C" void kernel_launch(void* const* d_in, const int* in_sizes, int n_in,
                              void* d_out, int out_size, void* d_ws, size_t ws_size,
                              hipStream_t stream) {
  (void)in_sizes; (void)n_in; (void)out_size; (void)ws_size;

  const float* x       = (const float*)d_in[0];
  const float* enc_out = (const float*)d_in[1];
  const float* V1      = (const float*)d_in[2];
  const float* comp1   = (const float*)d_in[3];
  const float* b1      = (const float*)d_in[4];
  const float* V2      = (const float*)d_in[5];
  const float* comp2   = (const float*)d_in[6];
  const float* b2      = (const float*)d_in[7];
  const float* V3      = (const float*)d_in[8];
  const float* comp3   = (const float*)d_in[9];
  const float* b3      = (const float*)d_in[10];
  const float* Wd      = (const float*)d_in[11];
  const float* bd      = (const float*)d_in[12];
  const float* Wg      = (const float*)d_in[13];
  const float* bg      = (const float*)d_in[14];
  const float* Wm1     = (const float*)d_in[15];
  const float* bm1     = (const float*)d_in[16];
  const float* Wm2     = (const float*)d_in[17];
  const float* bm2     = (const float*)d_in[18];
  const float* Wm3     = (const float*)d_in[19];
  const float* bm3     = (const float*)d_in[20];
  const int*   src     = (const int*)d_in[21];
  const int*   dst     = (const int*)d_in[22];
  const int*   etype   = (const int*)d_in[23];
  const int*   midx    = (const int*)d_in[24];
  const int*   mmask   = (const int*)d_in[25];
  float* out = (float*)d_out;

  // workspace carve-up
  char* ws = (char*)d_ws;
  size_t off = 0;
  auto take = [&](size_t bytes) -> char* {
    char* p = ws + off;
    off = alignup256(off + bytes);
    return p;
  };
  const size_t sz_wt1  = (size_t)R_ * H_ * INP_ * 2;        // [12][256][224] bf16
  const size_t sz_wt2  = (size_t)R_ * H_ * H_ * 2;          // [12][256][256] bf16
  const size_t sz_wt3  = (size_t)R_ * OUTP_ * H_ * 2;       // [12][208][256] bf16
  const size_t sz_wdt  = (size_t)OUTP_ * HS_ * 2;           // [208][768] bf16
  const size_t sz_xb1  = (size_t)NN_ * INP_ * 2;            // [16384][224] bf16
  const size_t sz_hb   = (size_t)NN_ * H_ * 2;              // [16384][256] bf16
  const size_t sz_encb = (size_t)(B_ * L_) * HS_ * 2;       // [16384][768] bf16
  const size_t sz_proj = (size_t)NN_ * (R_ * H_) * 2;       // [16384][3072] bf16 (layers reuse)
  const size_t sz_agg  = (size_t)NN_ * H_ * 4;              // [16384][256] f32
  const size_t sz_enc  = (size_t)(B_ * L_) * OUTP_ * 4;     // [16384][208] f32
  const size_t sz_gf   = (size_t)NN_ * OUT_ * 4;            // [16384][200] f32
  const size_t sz_deg  = (size_t)NN_ * 4;                   // int
  const size_t sz_offs = (size_t)NN_ * 4;                   // int
  const size_t sz_cur  = (size_t)NN_ * 4;                   // int
  const size_t sz_ebuf = (size_t)E_ * 4;                    // int

  u16*   wt1  = (u16*)take(sz_wt1);
  u16*   wt2  = (u16*)take(sz_wt2);
  u16*   wt3  = (u16*)take(sz_wt3);
  u16*   wdt  = (u16*)take(sz_wdt);
  u16*   xb1  = (u16*)take(sz_xb1);
  u16*   hb   = (u16*)take(sz_hb);
  u16*   encb = (u16*)take(sz_encb);
  u16*   proj = (u16*)take(sz_proj);
  float* agg  = (float*)take(sz_agg);
  float* enc  = (float*)take(sz_enc);
  float* gf   = (float*)take(sz_gf);
  int*   deg  = (int*)take(sz_deg);
  int*   offs = (int*)take(sz_offs);
  int*   cur  = (int*)take(sz_cur);
  int*   ebuf = (int*)take(sz_ebuf);

  const int T = 256;
  auto nblk = [](long total, int t) { return (unsigned)((total + t - 1) / t); };

  // zero padded buffers (pad lanes must be exactly 0 for the GEMMs) + CSR counters
  hipMemsetAsync(wt1, 0, sz_wt1, stream);
  hipMemsetAsync(wt3, 0, sz_wt3, stream);
  hipMemsetAsync(wdt, 0, sz_wdt, stream);
  hipMemsetAsync(xb1, 0, sz_xb1, stream);
  hipMemsetAsync(deg, 0, sz_deg, stream);
  hipMemsetAsync(cur, 0, sz_cur, stream);

  // ---- CSR over dst (shared by all three RGCN layers) ----
  deg_kernel<<<nblk(E_, T), T, 0, stream>>>(dst, deg);
  scan_kernel<<<1, 256, 0, stream>>>(deg, offs);
  fill_kernel<<<nblk(E_, T), T, 0, stream>>>(dst, offs, cur, ebuf);

  // ---- weight prep + input conversion ----
  prep_w_kernel<<<nblk((long)R_ * H_ * IN_, T), T, 0, stream>>>(V1, comp1, wt1, IN_, H_, INP_, H_);
  prep_w_kernel<<<nblk((long)R_ * H_ * H_, T), T, 0, stream>>>(V2, comp2, wt2, H_, H_, H_, H_);
  prep_w_kernel<<<nblk((long)R_ * OUT_ * H_, T), T, 0, stream>>>(V3, comp3, wt3, H_, OUT_, H_, OUTP_);
  prep_wd_kernel<<<nblk((long)OUT_ * HS_, T), T, 0, stream>>>(Wd, wdt);
  conv_pad_kernel<<<nblk((long)NN_ * IN_, T), T, 0, stream>>>(x, xb1, NN_ * IN_, IN_, INP_);
  conv_pad_kernel<<<nblk((long)(B_ * L_) * HS_, T), T, 0, stream>>>(enc_out, encb,
                                                                    (B_ * L_) * HS_, HS_, HS_);

  const dim3 gblk(256);

  // ---- RGCN layer 1: proj = xb1 @ W1_r -> CSR gather -> relu ----
  gemm_wmma_kernel<true><<<dim3(NN_ / 128, (R_ * H_) / 16), gblk, 0, stream>>>(
      xb1, wt1, proj, INP_, R_ * H_);
  node_agg_kernel<<<nblk((long)NN_ * 32, T), T, 0, stream>>>(proj, src, etype, offs, deg, ebuf,
                                                             agg, R_ * H_, H_, H_);
  bias_relu_bf16_kernel<<<nblk((long)NN_ * H_, T), T, 0, stream>>>(agg, b1, hb, NN_ * H_, H_);

  // ---- RGCN layer 2 ----
  gemm_wmma_kernel<true><<<dim3(NN_ / 128, (R_ * H_) / 16), gblk, 0, stream>>>(
      hb, wt2, proj, H_, R_ * H_);
  node_agg_kernel<<<nblk((long)NN_ * 32, T), T, 0, stream>>>(proj, src, etype, offs, deg, ebuf,
                                                             agg, R_ * H_, H_, H_);
  bias_relu_bf16_kernel<<<nblk((long)NN_ * H_, T), T, 0, stream>>>(agg, b2, hb, NN_ * H_, H_);

  // ---- RGCN layer 3 (no relu; bias folded into gf kernel) ----
  gemm_wmma_kernel<true><<<dim3(NN_ / 128, (R_ * OUTP_) / 16), gblk, 0, stream>>>(
      hb, wt3, proj, H_, R_ * OUTP_);
  node_agg_kernel<<<nblk((long)NN_ * 32, T), T, 0, stream>>>(proj, src, etype, offs, deg, ebuf,
                                                             agg, R_ * OUTP_, OUTP_, OUT_);

  // ---- encoder dense: enc = enc_out @ Wd (f32 out; bd added in gf kernel) ----
  gemm_wmma_kernel<false><<<dim3((B_ * L_) / 128, OUTP_ / 16), gblk, 0, stream>>>(
      encb, wdt, enc, HS_, OUTP_);

  // ---- gf = h3 + b3 + masked token mean (+bd) ----
  gf_fuse_kernel<<<nblk((long)NN_ * 32, T), T, 0, stream>>>(agg, b3, enc, bd, midx, mmask, gf);

  // ---- attention pooling + MLP head ----
  pool_mlp_kernel<<<B_, N_, 0, stream>>>(gf, Wg, bg, Wm1, bm1, Wm2, bm2, Wm3, bm3, out);
}
